// RQVAE_6554120093984
// MI455X (gfx1250) — matrix operand order repaired and checked by reading, not compiled, not run on hardware
//
#include <hip/hip_runtime.h>
#include <hip/hip_bf16.h>

// RQ-VAE forward for MI455X (gfx1250, wave32).
// GEMMs: v_wmma_f32_16x16x32_bf16, all operands pre-converted to bf16 in
// global memory (weights transposed once per layer), tiles moved with
// global_load_async_to_lds_b128 (ASYNCcnt), double-buffered LDS.

#define NROWS 16384
#define EDIM  64
#define NLEV  4
#define NCODE 256
#define BETA  0.25f

typedef __attribute__((ext_vector_type(16))) __bf16 v16bf;
typedef __attribute__((ext_vector_type(8)))  float  v8f;

#define BM 128
#define BN 128
#define BK 32
#define LDA 56   // LDS row stride in ushorts: 112B = 7*16 -> every 16B chunk aligned

union FragU { v16bf v; unsigned u[8]; };

__device__ __forceinline__ unsigned short f2bf(float f) {
  union { float f; unsigned u; } x; x.f = f;
  unsigned u = x.u;
  u += 0x7FFFu + ((u >> 16) & 1u);   // round-to-nearest-even
  return (unsigned short)(u >> 16);
}
__device__ __forceinline__ unsigned pack2(float a, float b) {
  return (unsigned)f2bf(a) | ((unsigned)f2bf(b) << 16);
}

// 16B global -> LDS async copy (ASYNCcnt-tracked). LDS offset = low 32 bits
// of the generic pointer to a __shared__ object.
__device__ __forceinline__ void async16(const unsigned short* g, unsigned short* l) {
  unsigned loff = (unsigned)(unsigned long long)l;
  asm volatile("global_load_async_to_lds_b128 %0, %1, off"
               :: "v"(loff), "v"(g) : "memory");
}
__device__ __forceinline__ void wait_async4() { asm volatile("s_wait_asynccnt 4" ::: "memory"); }
__device__ __forceinline__ void wait_async0() { asm volatile("s_wait_asynccnt 0" ::: "memory"); }

// C[N,M] = act(A[N,K] @ W[K,M] + bias). A bf16 [N][K], Wt bf16 [M][K]
// (pre-transposed). Writes fp32 (Cf) and/or bf16 (Cbf). K % 32 == 0.
__global__ __launch_bounds__(256) void gemm_bf16(
    const unsigned short* __restrict__ A, const unsigned short* __restrict__ Wt,
    const float* __restrict__ bias, float* __restrict__ Cf,
    unsigned short* __restrict__ Cbf, int K, int M, int relu)
{
  __shared__ __align__(16) unsigned short As[2][BM][LDA];
  __shared__ __align__(16) unsigned short Bs[2][BN][LDA];

  const int t    = threadIdx.x;
  const int lane = t & 31;
  const int wave = t >> 5;
  const int wr   = wave >> 1;    // 0..3 : 32-row group
  const int wc   = wave & 1;     // 0..1 : 64-col group
  const int r    = lane & 15;
  const int half = lane >> 4;

  const int m0 = blockIdx.y * BM;
  const int n0 = blockIdx.x * BN;

  v8f acc[2][4];
#pragma unroll
  for (int mb = 0; mb < 2; ++mb)
#pragma unroll
    for (int nb = 0; nb < 4; ++nb)
#pragma unroll
      for (int v = 0; v < 8; ++v) acc[mb][nb][v] = 0.0f;

  const int kTiles = K / BK;

  // Issue one 128x32 A tile + 128x32 B tile: 4 async b128 ops per thread.
  auto issue_tile = [&](int kt, int bufi) {
    const int k0 = kt * BK;
#pragma unroll
    for (int it = 0; it < 2; ++it) {
      int slot = it * 256 + t;
      int row  = slot >> 2;            // 0..127
      int c16  = (slot & 3) * 8;       // ushort offset: 0,8,16,24 (16B chunks)
      async16(A + (size_t)(m0 + row) * K + k0 + c16, &As[bufi][row][c16]);
    }
#pragma unroll
    for (int it = 0; it < 2; ++it) {
      int slot = it * 256 + t;
      int row  = slot >> 2;
      int c16  = (slot & 3) * 8;
      int gr   = n0 + row;
      if (gr >= M) gr = M - 1;         // clamp: keep ASYNCcnt uniform; junk cols discarded
      async16(Wt + (size_t)gr * K + k0 + c16, &Bs[bufi][row][c16]);
    }
  };

  issue_tile(0, 0);
  if (kTiles > 1) issue_tile(1, 1);

  for (int kt = 0; kt < kTiles; ++kt) {
    const int bufi = kt & 1;
    if (kt + 1 < kTiles) wait_async4(); else wait_async0();
    __syncthreads();

    FragU a[2], b[4];
#pragma unroll
    for (int mb = 0; mb < 2; ++mb) {
      const unsigned short* p = &As[bufi][wr * 32 + mb * 16 + r][half * 8];
      const unsigned* p0 = (const unsigned*)p;          // K = ks..ks+7
      const unsigned* p1 = (const unsigned*)(p + 16);   // K = ks+16..ks+23
#pragma unroll
      for (int i = 0; i < 4; ++i) { a[mb].u[i] = p0[i]; a[mb].u[4 + i] = p1[i]; }
    }
#pragma unroll
    for (int nb = 0; nb < 4; ++nb) {
      const unsigned* p = (const unsigned*)&Bs[bufi][wc * 64 + nb * 16 + r][half * 8];
#pragma unroll
      for (int i = 0; i < 4; ++i) { b[nb].u[i] = p[i]; b[nb].u[4 + i] = p[8 + i]; }
    }
#pragma unroll
    for (int mb = 0; mb < 2; ++mb)
#pragma unroll
      for (int nb = 0; nb < 4; ++nb)
        acc[mb][nb] = __builtin_amdgcn_wmma_f32_16x16x32_bf16(
            false, a[mb].v, false, b[nb].v, (short)0, acc[mb][nb], false, false);

    __syncthreads();                    // everyone done reading bufi
    if (kt + 2 < kTiles) issue_tile(kt + 2, bufi);
  }

  // Epilogue: bias (+ReLU); C layout: row = v + 8*half, col = lane&15.
#pragma unroll
  for (int mb = 0; mb < 2; ++mb) {
#pragma unroll
    for (int nb = 0; nb < 4; ++nb) {
      int col = n0 + wc * 64 + nb * 16 + r;
      if (col < M) {
        float bv = bias[col];
#pragma unroll
        for (int v = 0; v < 8; ++v) {
          int row = m0 + wr * 32 + mb * 16 + v + half * 8;
          float c = acc[mb][nb][v] + bv;
          if (relu) c = fmaxf(c, 0.0f);
          if (Cf)  Cf[(size_t)row * M + col]  = c;
          if (Cbf) Cbf[(size_t)row * M + col] = f2bf(c);
        }
      }
    }
  }
}

// Elementwise fp32 -> bf16 (n % 4 == 0).
__global__ __launch_bounds__(256) void act_to_bf16(
    const float* __restrict__ src, unsigned short* __restrict__ dst, long long n4)
{
  long long i = (long long)blockIdx.x * 256 + threadIdx.x;
  if (i < n4) {
    const float4 v = ((const float4*)src)[i];
    uint2 p; p.x = pack2(v.x, v.y); p.y = pack2(v.z, v.w);
    ((uint2*)dst)[i] = p;
  }
}

// W[K][M] fp32 -> Wt[M][K] bf16, 32x32 LDS tile transpose. K,M % 32 == 0.
__global__ __launch_bounds__(256) void wt_convert(
    const float* __restrict__ W, unsigned short* __restrict__ Wt, int K, int M)
{
  __shared__ unsigned short tile[32][36];
  const int t  = threadIdx.x;
  const int k0 = blockIdx.x * 32, m0 = blockIdx.y * 32;
  {
    int i = t >> 3, j4 = (t & 7) * 4;
    const float4 v = *(const float4*)(W + (size_t)(k0 + i) * M + m0 + j4);
    tile[i][j4 + 0] = f2bf(v.x); tile[i][j4 + 1] = f2bf(v.y);
    tile[i][j4 + 2] = f2bf(v.z); tile[i][j4 + 3] = f2bf(v.w);
  }
  __syncthreads();
  {
    int j = t >> 3, i4 = (t & 7) * 4;
    uint2 p;
    p.x = (unsigned)tile[i4 + 0][j] | ((unsigned)tile[i4 + 1][j] << 16);
    p.y = (unsigned)tile[i4 + 2][j] | ((unsigned)tile[i4 + 3][j] << 16);
    *(uint2*)(Wt + (size_t)(m0 + j) * K + k0 + i4) = p;
  }
}

// One block per row: 4-level residual VQ with straight-through output.
__global__ __launch_bounds__(256) void rq_vq_kernel(
    const float* __restrict__ xe, const float* __restrict__ cb,
    float* __restrict__ xq_out, unsigned short* __restrict__ xq_bf,
    float* __restrict__ idx_out, float* __restrict__ partials, int N)
{
  __shared__ float r[EDIM];
  __shared__ float sumq[EDIM];
  __shared__ float dist[NCODE];
  __shared__ int   didx[NCODE];
  __shared__ float sq[EDIM];

  const int n = blockIdx.x;
  const int t = threadIdx.x;

  if (t < EDIM) { r[t] = xe[(size_t)n * EDIM + t]; sumq[t] = 0.0f; }
  __syncthreads();

  for (int l = 0; l < NLEV; ++l) {
    const float* cbl = cb + (size_t)l * NCODE * EDIM;
    const float* c   = cbl + (size_t)t * EDIM;
    float d = 0.0f;
#pragma unroll 8
    for (int j = 0; j < EDIM; ++j) { float df = r[j] - c[j]; d += df * df; }
    dist[t] = d; didx[t] = t;
    __syncthreads();
    for (int s = 128; s > 0; s >>= 1) {
      if (t < s) {
        float d2 = dist[t + s]; int i2 = didx[t + s];
        if (d2 < dist[t] || (d2 == dist[t] && i2 < didx[t])) {
          dist[t] = d2; didx[t] = i2;
        }
      }
      __syncthreads();
    }
    const int best = didx[0];
    if (t < EDIM) {
      float q  = cbl[(size_t)best * EDIM + t];
      float df = q - r[t];
      sq[t]    = df * df;
      r[t]    -= q;
      sumq[t] += q;
    }
    __syncthreads();
    if (t < 32) sq[t] += sq[t + 32]; __syncthreads();
    if (t < 16) sq[t] += sq[t + 16]; __syncthreads();
    if (t <  8) sq[t] += sq[t +  8]; __syncthreads();
    if (t <  4) sq[t] += sq[t +  4]; __syncthreads();
    if (t <  2) sq[t] += sq[t +  2]; __syncthreads();
    if (t == 0) {
      partials[(size_t)l * N + n]   = sq[0] + sq[1];
      idx_out[(size_t)n * NLEV + l] = (float)best;
    }
    __syncthreads();
  }
  if (t < EDIM) {
    float v = sumq[t];
    xq_out[(size_t)n * EDIM + t] = v;          // fp32 copy for d_out
    xq_bf[(size_t)n * EDIM + t]  = f2bf(v);    // bf16 copy for decoder GEMM
  }
}

// Deterministic reduction of [NLEV][N] partials -> one scalar rq loss.
__global__ __launch_bounds__(256) void rq_loss_reduce(
    const float* __restrict__ partials, float* __restrict__ loss_out, int N)
{
  __shared__ float red[256];
  const int t = threadIdx.x;
  float total = 0.0f;
  for (int l = 0; l < NLEV; ++l) {
    float s = 0.0f;
    for (int i = t; i < N; i += 256) s += partials[(size_t)l * N + i];
    red[t] = s;
    __syncthreads();
    for (int st = 128; st > 0; st >>= 1) {
      if (t < st) red[t] += red[t + st];
      __syncthreads();
    }
    if (t == 0) total += (1.0f + BETA) * red[0] / (float)(N * EDIM);
    __syncthreads();
  }
  if (t == 0) *loss_out = total / (float)NLEV;
}

extern "C" void kernel_launch(void* const* d_in, const int* in_sizes, int n_in,
                              void* d_out, int out_size, void* d_ws, size_t ws_size,
                              hipStream_t stream) {
  (void)in_sizes; (void)n_in; (void)out_size; (void)ws_size;
  const int N = NROWS;

  // ---- workspace layout ----
  unsigned char* wsb = (unsigned char*)d_ws;
  unsigned short* pingbf = (unsigned short*)wsb;                       // N*2048 bf16
  unsigned short* pongbf = pingbf + (size_t)N * 2048;                  // N*2048 bf16
  float*          xe_f32 = (float*)(pongbf + (size_t)N * 2048);       // N*64 fp32
  unsigned short* xq_bf  = (unsigned short*)(xe_f32 + (size_t)N * EDIM); // N*64 bf16
  unsigned short* wtbuf  = xq_bf + (size_t)N * EDIM;                   // 4M bf16 (max W)
  float*          partials = (float*)(wtbuf + (size_t)4 * 1024 * 1024); // 3*NLEV*N fp32

  float* out_f = (float*)d_out;
  const size_t OUT_OFF[3] = {0, (size_t)N * 768, (size_t)N * 768 + (size_t)N * 1280};
  const size_t LOSS = (size_t)N * 768 + 2 * (size_t)N * 1280;
  const size_t IDX0 = LOSS + 3;
  const size_t XQ0  = IDX0 + 3 * (size_t)N * NLEV;

  static const int encdims[3][6] = {
      {768, 2048, 1024, 512, 256, 64},
      {1280, 2048, 1024, 512, 256, 64},
      {1280, 2048, 1024, 512, 256, 64}};
  static const int decdims[3][6] = {
      {64, 256, 512, 1024, 2048, 768},
      {64, 256, 512, 1024, 2048, 1280},
      {64, 256, 512, 1024, 2048, 1280}};

  for (int mod = 0; mod < 3; ++mod) {
    const int encbase = 6 + mod * 10;     // W0,b0,...,W4,b4 leaves
    const int decbase = 36 + mod * 10;
    const float* cb = (const float*)d_in[66 + mod];
    const int in_dim = encdims[mod][0];

    // ---- model input fp32 -> bf16 ----
    {
      long long n4 = (long long)N * in_dim / 4;
      act_to_bf16<<<dim3((unsigned)((n4 + 255) / 256)), 256, 0, stream>>>(
          (const float*)d_in[mod], pingbf, n4);
    }

    // ---- encoder ----
    const unsigned short* src = pingbf;
    for (int i = 0; i < 5; ++i) {
      const int Kd = encdims[mod][i], Md = encdims[mod][i + 1];
      wt_convert<<<dim3(Kd / 32, Md / 32), 256, 0, stream>>>(
          (const float*)d_in[encbase + 2 * i], wtbuf, Kd, Md);
      unsigned short* dstbf = (i == 4) ? nullptr : ((i & 1) ? pingbf : pongbf);
      float* dstf = (i == 4) ? xe_f32 : nullptr;
      dim3 g((Md + BN - 1) / BN, N / BM);
      gemm_bf16<<<g, 256, 0, stream>>>(
          src, wtbuf, (const float*)d_in[encbase + 2 * i + 1],
          dstf, dstbf, Kd, Md, (i < 4) ? 1 : 0);
      if (dstbf) src = dstbf;
    }

    // ---- residual VQ ----
    rq_vq_kernel<<<N, 256, 0, stream>>>(
        xe_f32, cb,
        out_f + XQ0 + (size_t)mod * N * EDIM,
        xq_bf,
        out_f + IDX0 + (size_t)mod * N * NLEV,
        partials + (size_t)mod * NLEV * N, N);
    rq_loss_reduce<<<1, 256, 0, stream>>>(
        partials + (size_t)mod * NLEV * N, out_f + LOSS + mod, N);

    // ---- decoder ----
    src = xq_bf;
    for (int i = 0; i < 5; ++i) {
      const int Kd = decdims[mod][i], Md = decdims[mod][i + 1];
      wt_convert<<<dim3(Kd / 32, Md / 32), 256, 0, stream>>>(
          (const float*)d_in[decbase + 2 * i], wtbuf, Kd, Md);
      unsigned short* dstbf = (i == 4) ? nullptr : ((i & 1) ? pongbf : pingbf);
      float* dstf = (i == 4) ? (out_f + OUT_OFF[mod]) : nullptr;
      dim3 g((Md + BN - 1) / BN, N / BM);
      gemm_bf16<<<g, 256, 0, stream>>>(
          src, wtbuf, (const float*)d_in[decbase + 2 * i + 1],
          dstf, dstbf, Kd, Md, (i < 4) ? 1 : 0);
      if (dstbf) src = dstbf;
    }
  }
}
